// DynaRepViT_79697413145230
// MI455X (gfx1250) — matrix operand order repaired and checked by reading, not compiled, not run on hardware
//
#include <hip/hip_runtime.h>
#include <math.h>

#define EPS 1e-5f

typedef __attribute__((ext_vector_type(16))) _Float16 v16h;
typedef __attribute__((ext_vector_type(8)))  float    v8f;

// ---------------------------------------------------------------------------
// Kernel 1: global average pool.  One wave32 per (b,c) plane of 56*56=3136.
// 768 blocks * 8 waves = 6144 planes.  Also warms L2 with x (77MB < 192MB L2).
// ---------------------------------------------------------------------------
__global__ __launch_bounds__(256) void pool_kernel(const float* __restrict__ x,
                                                   float* __restrict__ pooled)
{
    const int wave = threadIdx.x >> 5;
    const int lane = threadIdx.x & 31;
    const int p = blockIdx.x * 8 + wave;          // plane index < 6144
    const float4* xv = (const float4*)(x + (size_t)p * 3136);
    float s = 0.f;
    for (int i = lane; i < 784; i += 32) {        // 3136/4 float4s
        float4 v = xv[i];
        s += v.x + v.y + v.z + v.w;
    }
    #pragma unroll
    for (int m = 16; m >= 1; m >>= 1)
        s += __shfl_xor(s, m, 32);
    if (lane == 0) pooled[p] = s * (1.0f / 3136.0f);
}

// ---------------------------------------------------------------------------
// Kernel 2: kernel predictor.  pooled(64x96) @ kw1^T(96x4) via WMMA f16->f32,
// BN + exact GELU, 4->2 layer, per-sample branch selection.
// 1 block of 128 threads = 4 wave32s, one 16-row M-tile each.
// ---------------------------------------------------------------------------
__global__ __launch_bounds__(128) void predictor_kernel(
    const float* __restrict__ pooled,   // (64,96)
    const float* __restrict__ kw1,      // (4,96) row-major
    const float* __restrict__ kg1, const float* __restrict__ kb1,
    const float* __restrict__ km1, const float* __restrict__ kv1,
    const float* __restrict__ kw2,      // (2,4)
    const float* __restrict__ kg2, const float* __restrict__ kb2,
    const float* __restrict__ km2, const float* __restrict__ kv2,
    float* __restrict__ sel)            // (64,) 1.0 -> use 3x3 branch
{
    __shared__ float hbuf[64 * 4];
    const int wave = threadIdx.x >> 5;  // M tile: rows wave*16 .. +15
    const int lane = threadIdx.x & 31;
    const int mrow = lane & 15;         // M (and N for B/C) within tile
    const int hi   = lane >> 4;         // half-wave selector

    // branch-free B-fragment source: clamp row, mask value
    const float* brow  = kw1 + (mrow < 4 ? mrow : 0) * 96;
    const float  bmask = (mrow < 4) ? 1.0f : 0.0f;
    const float* prow  = pooled + (wave * 16 + mrow) * 96;

    v8f c = {};
    #pragma unroll
    for (int kb = 0; kb < 96; kb += 32) {
        v16h a, bm;
        #pragma unroll
        for (int e = 0; e < 16; ++e) {
            // A (16x32 f16): lane holds row M=lane%16; K layout per ISA 7.12.2
            int ka = 16 * (e >> 3) + 8 * hi + (e & 7);
            a[e] = (_Float16)prow[kb + ka];
            // B (32x16 f16): N = lane%16, K layout same as A
            int kbl = 16 * (e >> 3) + 8 * hi + (e & 7);
            bm[e] = (_Float16)(brow[kb + kbl] * bmask);
        }
        c = __builtin_amdgcn_wmma_f32_16x16x32_f16(
                false, a, false, bm, (short)0, c, false, false);
    }

    // C layout: c[r] holds (M = r + 8*hi, N = lane%16); keep N<4, BN+GELU.
    #pragma unroll
    for (int r = 0; r < 8; ++r) {
        int M = wave * 16 + r + 8 * hi;
        int N = mrow;
        if (N < 4) {
            float s1 = kg1[N] * rsqrtf(kv1[N] + EPS);
            float t1 = kb1[N] - km1[N] * s1;
            float h  = c[r] * s1 + t1;
            float g  = 0.5f * h * (1.0f + erff(h * 0.70710678118654752f));
            hbuf[M * 4 + N] = g;
        }
    }
    __syncthreads();

    if (threadIdx.x < 64) {
        int bi = threadIdx.x;
        float z0 = 0.f, z1 = 0.f;
        #pragma unroll
        for (int k = 0; k < 4; ++k) {
            float hv = hbuf[bi * 4 + k];
            z0 += hv * kw2[0 * 4 + k];
            z1 += hv * kw2[1 * 4 + k];
        }
        float s20 = kg2[0] * rsqrtf(kv2[0] + EPS);
        float s21 = kg2[1] * rsqrtf(kv2[1] + EPS);
        z0 = z0 * s20 + (kb2[0] - km2[0] * s20);
        z1 = z1 * s21 + (kb2[1] - km2[1] * s21);
        // softmax is monotone: kwts0 > kwts1  <=>  z0 > z1
        sel[bi] = (z0 > z1) ? 1.0f : 0.0f;
    }
}

// ---------------------------------------------------------------------------
// Kernel 3: fused depthwise conv + BN + select + 1x1 + final BN.
// One block per (b,c) plane.  Halo tile (stride 64, x index g -> col g+4) is
// DMA'd memory->LDS with gfx1250 global_load_async_to_lds_b128 (ASYNCcnt),
// border zeros are disjoint plain LDS stores.  Only the selected branch is
// computed (uniform per block):  out = A*conv + Wx*x + Cst.
// ---------------------------------------------------------------------------
__global__ __launch_bounds__(256) void fused_conv_kernel(
    const float* __restrict__ x,
    const float* __restrict__ w3, const float* __restrict__ g3,
    const float* __restrict__ b3, const float* __restrict__ m3,
    const float* __restrict__ v3,
    const float* __restrict__ w5, const float* __restrict__ g5,
    const float* __restrict__ b5, const float* __restrict__ m5,
    const float* __restrict__ v5,
    const float* __restrict__ w1, const float* __restrict__ b1,
    const float* __restrict__ gbn, const float* __restrict__ bbn,
    const float* __restrict__ mbn, const float* __restrict__ vbn,
    const float* __restrict__ sel,
    float* __restrict__ out)
{
    __shared__ __align__(16) float tile[64 * 60];   // row stride 64 floats
    const int p = blockIdx.x;            // b*96 + c
    const int b = p / 96;
    const int c = p % 96;
    const float* xp = x + (size_t)p * 3136;
    float* op = out + (size_t)p * 3136;

    // --- zero the border cells actually read (disjoint from DMA region) ---
    // rows 0,1,58,59 (full 64 cols) + cols {2,3,60,61} of rows 2..57.
    for (int i = threadIdx.x; i < 480; i += 256) {
        int idx;
        if (i < 256) {
            const int rmap[4] = {0, 1, 58, 59};
            idx = rmap[i >> 6] * 64 + (i & 63);
        } else {
            int k  = i - 256;                 // 0..223
            int rr = (k >> 2) + 2;            // rows 2..57
            int cc = k & 3;                   // {2,3,60,61}
            idx = rr * 64 + (cc < 2 ? 2 + cc : 58 + cc);
        }
        tile[idx] = 0.0f;
    }

    // --- async DMA: 56 rows x 14 x b128 (16B chunks), both sides 16B aligned.
    // global row r starts at xp + r*56 (224B stride); LDS dest col 4 => +16B.
    const unsigned ldsbase = (unsigned)(size_t)(void*)tile;
    for (int j = threadIdx.x; j < 784; j += 256) {
        int row = j / 14, c16 = j - row * 14;
        const float* g = xp + row * 56 + c16 * 4;
        unsigned l = ldsbase + (unsigned)((row + 2) * 256 + 16 + c16 * 16);
        asm volatile("global_load_async_to_lds_b128 %0, %1, off"
                     :: "v"(l), "v"(g) : "memory");
    }
    asm volatile("s_wait_asynccnt 0x0" ::: "memory");
    __syncthreads();

    const bool  use3 = sel[b] > 0.5f;
    const float sbn  = gbn[c] * rsqrtf(vbn[c] + EPS);
    const float tbn  = bbn[c] - mbn[c] * sbn;
    const float Wx   = sbn * w1[c];

    if (use3) {
        const float s3  = g3[c] * rsqrtf(v3[c] + EPS);
        const float A   = sbn * s3;
        const float Cst = sbn * ((b3[c] - m3[c] * s3) + b1[c]) + tbn;
        float w[9];
        #pragma unroll
        for (int i = 0; i < 9; ++i) w[i] = w3[c * 9 + i];
        for (int pix = threadIdx.x; pix < 3136; pix += 256) {
            int y = pix / 56, xx = pix - y * 56;
            const float* t = tile + (y + 2) * 64 + (xx + 4);
            float acc = 0.f;
            #pragma unroll
            for (int dy = -1; dy <= 1; ++dy)
                #pragma unroll
                for (int dx = -1; dx <= 1; ++dx)
                    acc = fmaf(t[dy * 64 + dx], w[(dy + 1) * 3 + (dx + 1)], acc);
            op[pix] = fmaf(A, acc, fmaf(Wx, t[0], Cst));
        }
    } else {
        const float s5  = g5[c] * rsqrtf(v5[c] + EPS);
        const float A   = sbn * s5;
        const float Cst = sbn * ((b5[c] - m5[c] * s5) + b1[c]) + tbn;
        float w[25];
        #pragma unroll
        for (int i = 0; i < 25; ++i) w[i] = w5[c * 25 + i];
        for (int pix = threadIdx.x; pix < 3136; pix += 256) {
            int y = pix / 56, xx = pix - y * 56;
            const float* t = tile + (y + 2) * 64 + (xx + 4);
            float acc = 0.f;
            #pragma unroll
            for (int dy = -2; dy <= 2; ++dy)
                #pragma unroll
                for (int dx = -2; dx <= 2; ++dx)
                    acc = fmaf(t[dy * 64 + dx], w[(dy + 2) * 5 + (dx + 2)], acc);
            op[pix] = fmaf(A, acc, fmaf(Wx, t[0], Cst));
        }
    }
}

// ---------------------------------------------------------------------------
extern "C" void kernel_launch(void* const* d_in, const int* in_sizes, int n_in,
                              void* d_out, int out_size, void* d_ws, size_t ws_size,
                              hipStream_t stream)
{
    (void)in_sizes; (void)n_in; (void)out_size; (void)ws_size;
    const float* x   = (const float*)d_in[0];
    const float* w3  = (const float*)d_in[1];
    const float* g3  = (const float*)d_in[2];
    const float* b3  = (const float*)d_in[3];
    const float* m3  = (const float*)d_in[4];
    const float* v3  = (const float*)d_in[5];
    const float* w5  = (const float*)d_in[6];
    const float* g5  = (const float*)d_in[7];
    const float* b5  = (const float*)d_in[8];
    const float* m5  = (const float*)d_in[9];
    const float* v5  = (const float*)d_in[10];
    const float* w1  = (const float*)d_in[11];
    const float* b1  = (const float*)d_in[12];
    const float* gbn = (const float*)d_in[13];
    const float* bbn = (const float*)d_in[14];
    const float* mbn = (const float*)d_in[15];
    const float* vbn = (const float*)d_in[16];
    const float* kw1 = (const float*)d_in[17];
    const float* kg1 = (const float*)d_in[18];
    const float* kb1 = (const float*)d_in[19];
    const float* km1 = (const float*)d_in[20];
    const float* kv1 = (const float*)d_in[21];
    const float* kw2 = (const float*)d_in[22];
    const float* kg2 = (const float*)d_in[23];
    const float* kb2 = (const float*)d_in[24];
    const float* km2 = (const float*)d_in[25];
    const float* kv2 = (const float*)d_in[26];

    float* pooled = (float*)d_ws;        // 6144 floats
    float* sel    = pooled + 6144;       // 64 floats

    pool_kernel<<<768, 256, 0, stream>>>(x, pooled);
    predictor_kernel<<<1, 128, 0, stream>>>(pooled, kw1, kg1, kb1, km1, kv1,
                                            kw2, kg2, kb2, km2, kv2, sel);
    fused_conv_kernel<<<6144, 256, 0, stream>>>(x,
        w3, g3, b3, m3, v3, w5, g5, b5, m5, v5, w1, b1,
        gbn, bbn, mbn, vbn, sel, (float*)d_out);
}
